// DyMuS_plus_39118562132049
// MI455X (gfx1250) — compile-verified, matrix operand order checked by reading, not run on hardware
//
#include <hip/hip_runtime.h>
#include <hip/hip_bf16.h>
#include <math.h>

// ---------------------------------------------------------------------------
// DyMuS+ on MI455X (gfx1250, wave32, WMMA).
// Heavy GEMMs use v_wmma_f32_16x16x32_bf16 (bf16 in, fp32 accumulate).
// All fragment loads are 16-byte vector loads (2 x b128 per fragment).
// gi = x@Wx is invariant across the 3 GRU calls -> computed once, kept bf16.
// ---------------------------------------------------------------------------

constexpr int Bb  = 128;          // batch
constexpr int Tt  = 50;           // seq len
constexpr int Mm  = 4;            // behaviors
constexpr int Dd  = 64;           // embed dim
constexpr int Kk  = 64;           // capsules
constexpr int Ll  = 16;           // capsule length
constexpr int Gg  = 3 * Kk * Ll;  // 3072 (GRU gate width)
constexpr int BT  = Bb * Tt;      // 6400
constexpr int NIT = 100001;       // num items + 1
constexpr int CAT = Dd + Ll;      // 80

typedef __attribute__((ext_vector_type(16))) __bf16 v16bf;
typedef __attribute__((ext_vector_type(8)))  __bf16 v8bf;
typedef __attribute__((ext_vector_type(4)))  __bf16 v4bf;
typedef __attribute__((ext_vector_type(8)))  float  v8f;

__device__ __forceinline__ int laneid() { return (int)(threadIdx.x & 31u); }

__device__ __forceinline__ v8f v8zero() {
  v8f c;
#pragma unroll
  for (int i = 0; i < 8; ++i) c[i] = 0.0f;
  return c;
}

__device__ __forceinline__ v16bf v16zero() {
  v16bf a;
#pragma unroll
  for (int i = 0; i < 16; ++i) a[i] = (__bf16)0.0f;
  return a;
}

__device__ __forceinline__ v16bf combine8(v8bf lo, v8bf hi) {
  v16bf a;
#pragma unroll
  for (int i = 0; i < 8; ++i) { a[i] = lo[i]; a[8 + i] = hi[i]; }
  return a;
}

// Per-lane fragment chunk loads: rowptr already = base + row*ld + kh*8.
// Elements k = {kh*8 .. kh*8+7} and {16+kh*8 .. 16+kh*8+7}: two b128 loads.
__device__ __forceinline__ v16bf load_frag16(const __bf16* rowptr) {
  v8bf lo = *(const v8bf*)(rowptr);
  v8bf hi = *(const v8bf*)(rowptr + 16);
  return combine8(lo, hi);
}

// A fragment (16x32 bf16) from row-major bf16 [rows, ld]; lane&15 = row.
__device__ __forceinline__ v16bf loadA_bf16(const __bf16* src, int ld) {
  int l = laneid();
  return load_frag16(src + (size_t)(l & 15) * ld + ((l >> 4) << 3));
}

// B fragment (32x16 bf16) from TRANSPOSED storage T[n][k] (row-major [N, ldT]).
__device__ __forceinline__ v16bf loadB_T(const __bf16* T, int ldT, int kbase) {
  int l = laneid();
  return load_frag16(T + (size_t)(l & 15) * ldT + kbase + ((l >> 4) << 3));
}

// Same with column guard (for the ragged item dimension).
__device__ __forceinline__ v16bf loadB_T_guard(const __bf16* T, int ldT, int kbase,
                                               int n0, int nmax) {
  int l = laneid();
  int n = n0 + (l & 15);
  if (n >= nmax) return v16zero();
  return load_frag16(T + (size_t)n * ldT + kbase + ((l >> 4) << 3));
}

// A fragment from row-major fp32 with K zero-padding (small operands only).
__device__ __forceinline__ v16bf loadA_f32(const float* src, int ld, int kvalid) {
  int l = laneid(), r = l & 15, kh = l >> 4;
  v16bf a;
#pragma unroll
  for (int i = 0; i < 16; ++i) {
    int k = ((i >> 3) << 4) + (kh << 3) + (i & 7);
    float v = (k < kvalid) ? src[(size_t)r * ld + k] : 0.0f;
    a[i] = (__bf16)v;
  }
  return a;
}

// B fragment, scalar path with K zero-padding (loop-invariant Wh only).
__device__ __forceinline__ v16bf loadB_f32_kv(const float* src, int ld, int kvalid) {
  int l = laneid(), n = l & 15, kh = l >> 4;
  v16bf b;
#pragma unroll
  for (int i = 0; i < 16; ++i) {
    int k = ((i >> 3) << 4) + (kh << 3) + (i & 7);
    b[i] = (k < kvalid) ? (__bf16)src[(size_t)k * ld + n] : (__bf16)0.0f;
  }
  return b;
}

// C/D fragment (16x16 fp32): VGPR v -> M = v + 8*(lane>>4), N = lane&15.
__device__ __forceinline__ v8f loadC_f32(const float* src, long ld) {
  int l = laneid(), n = l & 15, mo = (l >> 4) * 8;
  v8f c;
#pragma unroll
  for (int v = 0; v < 8; ++v) c[v] = src[(size_t)(v + mo) * ld + n];
  return c;
}

__device__ __forceinline__ v8f loadC_bf16(const __bf16* src, long ld) {
  int l = laneid(), n = l & 15, mo = (l >> 4) * 8;
  v8f c;
#pragma unroll
  for (int v = 0; v < 8; ++v) c[v] = (float)src[(size_t)(v + mo) * ld + n];
  return c;
}

__device__ __forceinline__ void storeC_f32(float* dst, long ld, v8f c) {
  int l = laneid(), n = l & 15, mo = (l >> 4) * 8;
#pragma unroll
  for (int v = 0; v < 8; ++v) dst[(size_t)(v + mo) * ld + n] = c[v];
}

__device__ __forceinline__ void storeC_bf16(__bf16* dst, long ld, v8f c) {
  int l = laneid(), n = l & 15, mo = (l >> 4) * 8;
#pragma unroll
  for (int v = 0; v < 8; ++v) dst[(size_t)(v + mo) * ld + n] = (__bf16)c[v];
}

__device__ __forceinline__ v8f wmma_bf16(v16bf a, v16bf b, v8f c) {
  return __builtin_amdgcn_wmma_f32_16x16x32_bf16(false, a, false, b, (short)0, c,
                                                 false, false);
}

__device__ __forceinline__ float sigmoidf_(float x) { return 1.0f / (1.0f + __expf(-x)); }

// ------------------------------- prep kernels ------------------------------

__global__ void k_cvt_bf16(const float* __restrict__ src, __bf16* __restrict__ dst,
                           size_t n4) {
  size_t i = (size_t)blockIdx.x * blockDim.x + threadIdx.x;
  size_t stride = (size_t)gridDim.x * blockDim.x;
  for (; i < n4; i += stride) {
    float4 s = ((const float4*)src)[i];
    v4bf o;
    o[0] = (__bf16)s.x; o[1] = (__bf16)s.y; o[2] = (__bf16)s.z; o[3] = (__bf16)s.w;
    *(v4bf*)&dst[i * 4] = o;
  }
}

__global__ void k_zero_f32(float* __restrict__ dst, size_t n4) {
  size_t i = (size_t)blockIdx.x * blockDim.x + threadIdx.x;
  size_t stride = (size_t)gridDim.x * blockDim.x;
  float4 z = make_float4(0.f, 0.f, 0.f, 0.f);
  for (; i < n4; i += stride) ((float4*)dst)[i] = z;
}

// WxT[m][g][d] = Wx[m][d][g]  (one-time 3MB transpose so B loads vectorize)
__global__ void k_wxT(const float* __restrict__ Wx, __bf16* __restrict__ WxT) {
  int idx = blockIdx.x * blockDim.x + threadIdx.x;
  if (idx >= Mm * Dd * Gg) return;
  int d = idx & 63;
  int g = (idx >> 6) % Gg;
  int m = idx / (Dd * Gg);
  WxT[(size_t)(m * Gg + g) * Dd + d] = (__bf16)Wx[(size_t)(m * Dd + d) * Gg + g];
}

// Embed + mask -> x_bf16 laid out [M, B*T, D]; float4 in, v4bf out.
__global__ void k_embed(const int* __restrict__ s0, const int* __restrict__ s1,
                        const int* __restrict__ s2, const int* __restrict__ s3,
                        const float* __restrict__ emb, __bf16* __restrict__ xbf) {
  int idx = blockIdx.x * blockDim.x + threadIdx.x;   // Mm*BT*16 threads
  if (idx >= Mm * BT * (Dd / 4)) return;
  int c4 = idx & 15;
  int row = idx >> 4;                                // m*BT + b*Tt + t
  int t = row % Tt;
  int tmp = row / Tt;
  int b = tmp % Bb;
  int m = tmp / Bb;
  const int* seqs[4] = {s0, s1, s2, s3};
  int item = seqs[m][b * Tt + t];
  float4 e = (item != 0) ? ((const float4*)emb)[(size_t)item * (Dd / 4) + c4]
                         : make_float4(0.f, 0.f, 0.f, 0.f);
  v4bf o;
  o[0] = (__bf16)e.x; o[1] = (__bf16)e.y; o[2] = (__bf16)e.z; o[3] = (__bf16)e.w;
  *(v4bf*)&xbf[(size_t)row * Dd + c4 * 4] = o;
}

// ----------------------- gi = x @ Wx + bx (WMMA GEMM) ----------------------
// Per m: [BT,64] @ [64,3072]. One wave: 16x64 tile, 8 WMMAs, all-b128 loads.
__global__ void k_gi_gemm(const __bf16* __restrict__ xbf, const __bf16* __restrict__ WxT,
                          const float* __restrict__ bx, __bf16* __restrict__ gi) {
  int rt = blockIdx.x;   // row tile (400)
  int nc = blockIdx.y;   // 64-col chunk (48)
  int m  = blockIdx.z;   // behavior (4)
  const __bf16* xm  = xbf + (size_t)m * BT * Dd + (size_t)rt * 16 * Dd;
  const __bf16* wmT = WxT + (size_t)m * Gg * Dd;
  v16bf a0 = loadA_bf16(xm, Dd);       // K 0..31
  v16bf a1 = loadA_bf16(xm + 32, Dd);  // K 32..63
  int lanen = laneid() & 15;
#pragma unroll
  for (int s = 0; s < 4; ++s) {
    int n0 = nc * 64 + s * 16;
    const __bf16* bT = wmT + (size_t)n0 * Dd;
    v16bf b0 = loadB_T(bT, Dd, 0);
    v16bf b1 = loadB_T(bT, Dd, 32);
    v8f c = v8zero();
    c = wmma_bf16(a0, b0, c);
    c = wmma_bf16(a1, b1, c);
    float bias = bx[m * Gg + n0 + lanen];
#pragma unroll
    for (int v = 0; v < 8; ++v) c[v] += bias;
    storeC_bf16(gi + (size_t)m * BT * Gg + (size_t)rt * 16 * Gg + n0, Gg, c);
  }
}

// ------------------------------ dynamic GRU --------------------------------
// One wave per (b-tile of 16, k, m). h kept in LDS *in A-fragment layout*
// (bf16, lane-contiguous) so the per-step A load is one ds_load_b128/lane.
__global__ void k_gru(const __bf16* __restrict__ gi, const float* __restrict__ Cbuf,
                      const float* __restrict__ Wh, const float* __restrict__ bh,
                      float* __restrict__ hfin, __bf16* __restrict__ behN, int writeN) {
  int bt = blockIdx.x;   // 8
  int k  = blockIdx.y;   // 64
  int m  = blockIdx.z;   // 4
  int b0 = bt * 16;
  int lane = laneid();
  int ncol = lane & 15, mo = (lane >> 4) * 8;

  // h in A-fragment swizzled layout: lane la = row + 16*(l>>3) holds
  // elements h[row][ (l>>3)*8 + 0..7 ] contiguously at habuf[la*16 ..].
  __shared__ __attribute__((aligned(16))) __bf16 habuf[32 * 16];
  for (int i = threadIdx.x; i < 512; i += 32) habuf[i] = (__bf16)0.0f;
  __syncthreads();

  // Wh[m,k] : [L=16, 48] -> 3 gate B-fragments (K padded 16->32), loop invariant.
  const float* wh = Wh + (size_t)(m * Kk + k) * Ll * 48;
  v16bf Bg0 = loadB_f32_kv(wh +  0, 48, Ll);
  v16bf Bg1 = loadB_f32_kv(wh + 16, 48, Ll);
  v16bf Bg2 = loadB_f32_kv(wh + 32, 48, Ll);
  float bh0 = bh[(m * Kk + k) * 48 +  0 + ncol];
  float bh1 = bh[(m * Kk + k) * 48 + 16 + ncol];
  float bh2 = bh[(m * Kk + k) * 48 + 32 + ncol];

  v8f hc = v8zero();

  const long giLd = (long)Tt * Gg;              // b-stride in gi rows
  const long cLd  = (long)Mm * Tt * Kk * Ll;    // b-stride in [B,M,T,K,L]
  const __bf16* giM = gi + (size_t)m * BT * Gg;

  // A-layout write targets for this lane's 8 C-fragment values:
  // value v holds h[v+mo][ncol] -> lane (v+mo) + 16*(ncol>>3), elem ncol&7.
  int wbase = 16 * (ncol >> 3) * 16 + (ncol & 7);

  for (int t = 0; t < Tt; ++t) {
    v8bf lo = *(const v8bf*)&habuf[lane * 16];   // one b128 LDS load
    v16bf a = combine8(lo, (v8bf)(__bf16)0.0f);  // K 16..31 zero-padded
    v8f gh0 = wmma_bf16(a, Bg0, v8zero());
    v8f gh1 = wmma_bf16(a, Bg1, v8zero());
    v8f gh2 = wmma_bf16(a, Bg2, v8zero());

    const __bf16* gip = giM + ((size_t)b0 * Tt + t) * Gg;
    v8f gi0 = loadC_bf16(gip + 0 * Kk * Ll + k * Ll, giLd);
    v8f gi1 = loadC_bf16(gip + 1 * Kk * Ll + k * Ll, giLd);
    v8f gi2 = loadC_bf16(gip + 2 * Kk * Ll + k * Ll, giLd);
    size_t cOff = (((size_t)b0 * Mm + m) * Tt + t) * Kk * Ll + (size_t)k * Ll;
    v8f Ct = loadC_f32(Cbuf + cOff, cLd);

    v8f nn;
#pragma unroll
    for (int v = 0; v < 8; ++v) {
      float r = sigmoidf_(gi0[v] + gh0[v] + bh0);
      float z = sigmoidf_(gi1[v] + gh1[v] + bh1 + Ct[v]);
      float n = tanhf(gi2[v] + r * (gh2[v] + bh2));
      nn[v] = n;
      hc[v] = (1.0f - z) * n + z * hc[v];
    }
    if (writeN) storeC_bf16(behN + cOff, cLd, nn);

    __syncthreads();
#pragma unroll
    for (int v = 0; v < 8; ++v) habuf[(v + mo) * 16 + wbase] = (__bf16)hc[v];
    __syncthreads();
  }
  // hfin: [B,M,K,L]
  storeC_f32(hfin + ((size_t)b0 * Mm + m) * Kk * Ll + (size_t)k * Ll,
             (long)Mm * Kk * Ll, hc);
}

// -------------------- routing elementwise / small kernels ------------------

// u[b,c,C,d] = sum_t hfin[b,t,c,d] * W[c,d,t,C]    (t = behavior, 4 MACs)
__global__ void k_u(const float* __restrict__ hfin, const float* __restrict__ W,
                    float* __restrict__ u) {
  size_t idx = (size_t)blockIdx.x * blockDim.x + threadIdx.x;
  if (idx >= (size_t)Bb * Kk * Kk * Ll) return;
  int d = (int)(idx & 15);
  size_t r = idx >> 4;
  int Cc = (int)(r & 63); r >>= 6;
  int c  = (int)(r & 63);
  int b  = (int)(r >> 6);
  float acc = 0.0f;
#pragma unroll
  for (int t = 0; t < Mm; ++t)
    acc += hfin[(((size_t)b * Mm + t) * Kk + c) * Ll + d] *
           W[(((size_t)c * Ll + d) * Mm + t) * Kk + Cc];
  u[idx] = acc;
}

// csm[b,i,:] = softmax_j(bmat[b,i,:])
__global__ void k_csoftmax(const float* __restrict__ bmat, float* __restrict__ csm) {
  int b = blockIdx.x, i = blockIdx.y, j = threadIdx.x;
  __shared__ float red[64];
  float x = bmat[((size_t)b * Kk + i) * Kk + j];
  red[j] = x; __syncthreads();
  for (int s = 32; s > 0; s >>= 1) { if (j < s) red[j] = fmaxf(red[j], red[j + s]); __syncthreads(); }
  float mx = red[0]; __syncthreads();
  float e = __expf(x - mx);
  red[j] = e; __syncthreads();
  for (int s = 32; s > 0; s >>= 1) { if (j < s) red[j] += red[j + s]; __syncthreads(); }
  csm[((size_t)b * Kk + i) * Kk + j] = e / red[0];
}

// v[b,j,d] = alpha * sum_i csm[b,i,j] * u[b,i,j,d]
__global__ void k_v(const float* __restrict__ csm, const float* __restrict__ u,
                    const float* __restrict__ alpha, float* __restrict__ v) {
  int idx = blockIdx.x * blockDim.x + threadIdx.x;
  if (idx >= Bb * Kk * Ll) return;
  int d = idx & 15, j = (idx >> 4) & 63, b = idx >> 10;
  float acc = 0.0f;
  for (int i = 0; i < Kk; ++i)
    acc += csm[((size_t)b * Kk + i) * Kk + j] *
           u[(((size_t)b * Kk + i) * Kk + j) * Ll + d];
  v[idx] = alpha[0] * acc;
}

// vq[b,j] = ||v[b,j,:]|| * w[j] + bias[j]
__global__ void k_vnorm(const float* __restrict__ v, const float* __restrict__ w,
                        const float* __restrict__ bias, float* __restrict__ vq) {
  int idx = blockIdx.x * blockDim.x + threadIdx.x;
  if (idx >= Bb * Kk) return;
  int j = idx & 63;
  float s = 0.0f;
#pragma unroll
  for (int d = 0; d < Ll; ++d) { float x = v[(size_t)idx * Ll + d]; s += x * x; }
  vq[idx] = sqrtf(s) * w[j] + bias[j];
}

// out[b,n] = sum_k vq[b,k] * E[n,k]   (WMMA; E rows are already B^T layout)
__global__ void k_logits(const float* __restrict__ vq, const __bf16* __restrict__ EBF,
                         float* __restrict__ out) {
  int nt = blockIdx.x;   // 64-col chunk
  int mt = blockIdx.y;   // 16-row chunk (8)
  const float* arow = vq + (size_t)mt * 16 * Kk;
  v16bf a0 = loadA_f32(arow, Kk, 32);
  v16bf a1 = loadA_f32(arow + 32, Kk, 32);
  int lane = laneid();
  int lanen = lane & 15, mo = (lane >> 4) * 8;
#pragma unroll
  for (int s = 0; s < 4; ++s) {
    int n0 = nt * 64 + s * 16;
    if (n0 >= NIT) return;          // wave-uniform
    v16bf b0 = loadB_T_guard(EBF, Dd, 0,  n0, NIT);
    v16bf b1 = loadB_T_guard(EBF, Dd, 32, n0, NIT);
    v8f c = v8zero();
    c = wmma_bf16(a0, b0, c);
    c = wmma_bf16(a1, b1, c);
    int n = n0 + lanen;
    if (n < NIT) {
#pragma unroll
      for (int v = 0; v < 8; ++v) out[(size_t)(mt * 16 + v + mo) * NIT + n] = c[v];
    }
  }
}

// Per-row max and sum(exp) of logits.
__global__ void k_smstats(const float* __restrict__ logits, float* __restrict__ rowmax,
                          float* __restrict__ rowsum) {
  int b = blockIdx.x, tid = threadIdx.x;
  __shared__ float red[256];
  const float* row = logits + (size_t)b * NIT;
  float mx = -3.4e38f;
  for (int n = tid; n < NIT; n += 256) mx = fmaxf(mx, row[n]);
  red[tid] = mx; __syncthreads();
  for (int s = 128; s > 0; s >>= 1) { if (tid < s) red[tid] = fmaxf(red[tid], red[tid + s]); __syncthreads(); }
  mx = red[0]; __syncthreads();
  float sm = 0.0f;
  for (int n = tid; n < NIT; n += 256) sm += __expf(row[n] - mx);
  red[tid] = sm; __syncthreads();
  for (int s = 128; s > 0; s >>= 1) { if (tid < s) red[tid] += red[tid + s]; __syncthreads(); }
  if (tid == 0) { rowmax[b] = mx; rowsum[b] = red[0]; }
}

// p[b,d] = sum_n softmax(logits)[b,n] * E[n,d]
__global__ void k_p(const float* __restrict__ logits, const float* __restrict__ E,
                    const float* __restrict__ rowmax, const float* __restrict__ rowsum,
                    float* __restrict__ p) {
  int b = blockIdx.x, tid = threadIdx.x;
  int d = tid & 63, w = tid >> 6;  // 4 groups x 64 lanes
  __shared__ float red[256];
  float mx = rowmax[b], inv = 1.0f / rowsum[b];
  const float* row = logits + (size_t)b * NIT;
  float acc = 0.0f;
  for (int n = w; n < NIT; n += 4) acc += __expf(row[n] - mx) * E[(size_t)n * Dd + d];
  red[tid] = acc; __syncthreads();
  if (w == 0) p[(size_t)b * Dd + d] = (red[d] + red[64 + d] + red[128 + d] + red[192 + d]) * inv;
}

// r[b,c,D] = sum_j cat(p[b],v[b,c])[j] * Wcoef[c,j,D]
__global__ void k_r(const float* __restrict__ p, const float* __restrict__ v,
                    const float* __restrict__ Wcoef, float* __restrict__ r) {
  int idx = blockIdx.x * blockDim.x + threadIdx.x;
  if (idx >= Bb * Kk * Ll) return;
  int Dl = idx & 15, c = (idx >> 4) & 63, b = idx >> 10;
  float acc = 0.0f;
  for (int j = 0; j < CAT; ++j) {
    float xv = (j < Dd) ? p[(size_t)b * Dd + j]
                        : v[((size_t)b * Kk + c) * Ll + (j - Dd)];
    acc += xv * Wcoef[((size_t)c * CAT + j) * Ll + Dl];
  }
  r[idx] = acc;
}

// bmat[b,i,j] += sum_d u[b,i,j,d] * r[b,j,d]
__global__ void k_bupd(const float* __restrict__ u, const float* __restrict__ r,
                       float* __restrict__ bmat) {
  int idx = blockIdx.x * blockDim.x + threadIdx.x;
  if (idx >= Bb * Kk * Kk) return;
  int j = idx & 63, b = idx >> 12;
  float acc = 0.0f;
#pragma unroll
  for (int d = 0; d < Ll; ++d)
    acc += u[(size_t)idx * Ll + d] * r[((size_t)b * Kk + j) * Ll + d];
  bmat[idx] += acc;
}

// C[b,m,t,k,l] += behN[b,m,t,k,l] * r[b,k,l]   (float4 / v4bf vectorized)
__global__ void k_cupd(const __bf16* __restrict__ behN, const float* __restrict__ r,
                       float* __restrict__ C) {
  size_t total4 = (size_t)Bb * Mm * Tt * Kk * Ll / 4;
  size_t i4 = (size_t)blockIdx.x * blockDim.x + threadIdx.x;
  size_t stride = (size_t)gridDim.x * blockDim.x;
  for (; i4 < total4; i4 += stride) {
    size_t idx = i4 * 4;
    int l0 = (int)(idx & 15);
    size_t q = idx >> 4;
    int k = (int)(q & 63);
    int b = (int)((q >> 6) / (Mm * Tt));
    float4 rv = *(const float4*)&r[((size_t)b * Kk + k) * Ll + l0];
    float4 cv = *(float4*)&C[idx];
    v4bf nv = *(const v4bf*)&behN[idx];
    cv.x += (float)nv[0] * rv.x;
    cv.y += (float)nv[1] * rv.y;
    cv.z += (float)nv[2] * rv.z;
    cv.w += (float)nv[3] * rv.w;
    *(float4*)&C[idx] = cv;
  }
}

// ------------------------------- launcher ----------------------------------

extern "C" void kernel_launch(void* const* d_in, const int* in_sizes, int n_in,
                              void* d_out, int out_size, void* d_ws, size_t ws_size,
                              hipStream_t stream) {
  const int*   purchase = (const int*)d_in[0];
  const int*   cart     = (const int*)d_in[1];
  const int*   fav      = (const int*)d_in[2];
  const int*   click    = (const int*)d_in[3];
  const float* item_emb = (const float*)d_in[4];
  const float* W        = (const float*)d_in[5];
  const float* W_coef   = (const float*)d_in[6];
  const float* alpha    = (const float*)d_in[7];
  const float* wvec     = (const float*)d_in[8];
  const float* biasv    = (const float*)d_in[9];
  const float* gru_Wx   = (const float*)d_in[10];
  const float* gru_bx   = (const float*)d_in[11];
  const float* gru_Wh   = (const float*)d_in[12];
  const float* gru_bh   = (const float*)d_in[13];
  float* out = (float*)d_out;

  // Workspace carving (~424 MB total).
  char* base = (char*)d_ws;
  size_t off = 0;
  auto carve = [&](size_t bytes) -> char* {
    off = (off + 255) & ~(size_t)255;
    char* p = base + off;
    off += bytes;
    return p;
  };
  __bf16* xbf    = (__bf16*)carve((size_t)Mm * BT * Dd * 2);
  __bf16* WxT    = (__bf16*)carve((size_t)Mm * Gg * Dd * 2);
  __bf16* EBF    = (__bf16*)carve((size_t)NIT * Dd * 2);
  __bf16* gi     = (__bf16*)carve((size_t)Mm * BT * Gg * 2);
  __bf16* behN   = (__bf16*)carve((size_t)Bb * Mm * Tt * Kk * Ll * 2);
  float*  Cbuf   = (float*)carve((size_t)Bb * Mm * Tt * Kk * Ll * 4);
  float*  hfin   = (float*)carve((size_t)Bb * Mm * Kk * Ll * 4);
  float*  ubuf   = (float*)carve((size_t)Bb * Kk * Kk * Ll * 4);
  float*  bmat   = (float*)carve((size_t)Bb * Kk * Kk * 4);
  float*  csm    = (float*)carve((size_t)Bb * Kk * Kk * 4);
  float*  vbuf   = (float*)carve((size_t)Bb * Kk * Ll * 4);
  float*  vq     = (float*)carve((size_t)Bb * Kk * 4);
  float*  logits = (float*)carve((size_t)Bb * NIT * 4);
  float*  rowmax = (float*)carve((size_t)Bb * 4);
  float*  rowsum = (float*)carve((size_t)Bb * 4);
  float*  pbuf   = (float*)carve((size_t)Bb * Dd * 4);
  float*  rbuf   = (float*)carve((size_t)Bb * Kk * Ll * 4);
  (void)ws_size; (void)n_in; (void)in_sizes; (void)out_size;

  // ---- prep: bf16 conversions / transpose, zero-init, embedding ----
  k_wxT<<<(Mm * Dd * Gg + 255) / 256, 256, 0, stream>>>(gru_Wx, WxT);
  k_cvt_bf16<<<2048, 256, 0, stream>>>(item_emb, EBF, (size_t)NIT * Dd / 4);
  k_zero_f32<<<4096, 256, 0, stream>>>(Cbuf, (size_t)Bb * Mm * Tt * Kk * Ll / 4);
  k_zero_f32<<<512, 256, 0, stream>>>(bmat, (size_t)Bb * Kk * Kk / 4);
  k_embed<<<(Mm * BT * (Dd / 4) + 255) / 256, 256, 0, stream>>>(
      purchase, cart, fav, click, item_emb, xbf);

  // ---- gi = x @ Wx + bx : invariant across all 3 GRU calls ----
  k_gi_gemm<<<dim3(BT / 16, Gg / 64, Mm), 32, 0, stream>>>(xbf, WxT, gru_bx, gi);

  // ---- 2 routing iterations ----
  for (int it = 0; it < 2; ++it) {
    k_gru<<<dim3(Bb / 16, Kk, Mm), 32, 0, stream>>>(gi, Cbuf, gru_Wh, gru_bh,
                                                    hfin, behN, 1);
    k_u<<<(int)(((size_t)Bb * Kk * Kk * Ll + 255) / 256), 256, 0, stream>>>(hfin, W, ubuf);
    k_csoftmax<<<dim3(Bb, Kk), Kk, 0, stream>>>(bmat, csm);
    k_v<<<(Bb * Kk * Ll + 255) / 256, 256, 0, stream>>>(csm, ubuf, alpha, vbuf);
    k_vnorm<<<(Bb * Kk + 255) / 256, 256, 0, stream>>>(vbuf, wvec, biasv, vq);
    k_logits<<<dim3((NIT + 63) / 64, Bb / 16), 32, 0, stream>>>(vq, EBF, logits);
    k_smstats<<<Bb, 256, 0, stream>>>(logits, rowmax, rowsum);
    k_p<<<Bb, 256, 0, stream>>>(logits, item_emb, rowmax, rowsum, pbuf);
    k_r<<<(Bb * Kk * Ll + 255) / 256, 256, 0, stream>>>(pbuf, vbuf, W_coef, rbuf);
    k_bupd<<<(Bb * Kk * Kk + 255) / 256, 256, 0, stream>>>(ubuf, rbuf, bmat);
    k_cupd<<<8192, 256, 0, stream>>>(behN, rbuf, Cbuf);
  }

  // ---- final pass: GRU -> u_hat -> v -> v_ -> logits into d_out ----
  k_gru<<<dim3(Bb / 16, Kk, Mm), 32, 0, stream>>>(gi, Cbuf, gru_Wh, gru_bh,
                                                  hfin, behN, 0);
  k_u<<<(int)(((size_t)Bb * Kk * Kk * Ll + 255) / 256), 256, 0, stream>>>(hfin, W, ubuf);
  k_csoftmax<<<dim3(Bb, Kk), Kk, 0, stream>>>(bmat, csm);
  k_v<<<(Bb * Kk * Ll + 255) / 256, 256, 0, stream>>>(csm, ubuf, alpha, vbuf);
  k_vnorm<<<(Bb * Kk + 255) / 256, 256, 0, stream>>>(vbuf, wvec, biasv, vq);
  k_logits<<<dim3((NIT + 63) / 64, Bb / 16), 32, 0, stream>>>(vq, EBF, out);
}
